// DecoderRNN_87265145520381
// MI455X (gfx1250) — matrix operand order repaired and checked by reading, not compiled, not run on hardware
//
#include <hip/hip_runtime.h>
#include <cmath>

// ---------------------------------------------------------------------------
// Decoder RNN step (attention + GRU + LM head) for MI455X / gfx1250.
// Big GEMMs run on v_wmma_f32_16x16x32_bf16 (f32 accumulate), everything else
// is fused elementwise. wave32, 256-thread blocks (8 waves / block).
// ---------------------------------------------------------------------------

#define N_BATCH 256
#define S_ENC   128
#define H_DIM   1024
#define E_DIM   512
#define V_DIM   50257
#define V_PAD   50272   // V rounded up to multiple of 16 (and of 32 for transpose tiles)

typedef __attribute__((ext_vector_type(16))) __bf16          bf16x16;
typedef __attribute__((ext_vector_type(16))) unsigned short  u16x16;
typedef __attribute__((ext_vector_type(8)))  float           f32x8;

// float -> bf16 bits, round-to-nearest-even
__device__ __forceinline__ unsigned short f2bf(float f) {
    unsigned u = __float_as_uint(f);
    u += 0x7fffu + ((u >> 16) & 1u);
    return (unsigned short)(u >> 16);
}

__device__ __forceinline__ f32x8 wmma_bf16(bf16x16 a, bf16x16 b, f32x8 c) {
    // 8 args: (neg_a, A, neg_b, B, c_mod, C, reuse_a, reuse_b)
    return __builtin_amdgcn_wmma_f32_16x16x32_bf16(false, a, false, b, (short)0, c, false, false);
}

// Build the per-lane 16-bit A fragment (16x32 MxK tile) from a row-major f32
// row pointer already offset by (lane>=16 ? 16 : 0) in K.
// Layout: lanes 0-15 -> elems 0..15 = K0..15
//         lanes 16-31 -> elems 0..15 = K16..31
__device__ __forceinline__ bf16x16 pack_a_frag(const float* __restrict__ ap) {
    float4 a0 = *(const float4*)(ap);
    float4 a1 = *(const float4*)(ap + 4);
    float4 a2 = *(const float4*)(ap + 8);
    float4 a3 = *(const float4*)(ap + 12);
    u16x16 t;
    t[0]  = f2bf(a0.x); t[1]  = f2bf(a0.y); t[2]  = f2bf(a0.z); t[3]  = f2bf(a0.w);
    t[4]  = f2bf(a1.x); t[5]  = f2bf(a1.y); t[6]  = f2bf(a1.z); t[7]  = f2bf(a1.w);
    t[8]  = f2bf(a2.x); t[9]  = f2bf(a2.y); t[10] = f2bf(a2.z); t[11] = f2bf(a2.w);
    t[12] = f2bf(a3.x); t[13] = f2bf(a3.y); t[14] = f2bf(a3.z); t[15] = f2bf(a3.w);
    return __builtin_bit_cast(bf16x16, t);
}

// ---------------------------------------------------------------------------
// Weight transpose + f32->bf16 convert: W (K x Nc, row-major f32) ->
// Wt (Ncp x K, row-major bf16 bits). Rows n >= Nc are zero-filled.
// grid = (Ncp/32, K/32), block = (32,8). K, Ncp multiples of 32.
// ---------------------------------------------------------------------------
__global__ __launch_bounds__(256)
void convT_kernel(const float* __restrict__ W, unsigned short* __restrict__ Wt,
                  int K, int Nc, int Ncp) {
    __shared__ float tile[32][33];
    int n0 = blockIdx.x * 32;
    int k0 = blockIdx.y * 32;
    int tx = threadIdx.x, ty = threadIdx.y;
    #pragma unroll
    for (int i = ty; i < 32; i += 8) {
        int n = n0 + tx;
        tile[i][tx] = (n < Nc) ? W[(size_t)(k0 + i) * Nc + n] : 0.0f;
    }
    __syncthreads();
    #pragma unroll
    for (int i = ty; i < 32; i += 8) {
        Wt[(size_t)(n0 + i) * K + (k0 + tx)] = f2bf(tile[tx][i]);
    }
}

// ---------------------------------------------------------------------------
// Generic bf16 WMMA GEMM: C(M x Nc) = act(A(M x K, f32) @ Bt^T + bias)
// Bt is (Ncp x K) bf16 (transposed weights).  ACT: 0 = none, 1 = relu.
// ACCUM: accumulators initialized from C (for two-term pre-activations).
// grid = (ceil(Ncp/512), M/16), block = 256 (8 waves, 4 col-tiles each).
// ---------------------------------------------------------------------------
template<int ACT, bool ACCUM>
__global__ __launch_bounds__(256)
void gemm_bf16_kernel(const float* __restrict__ A, const unsigned short* __restrict__ Bt,
                      const float* __restrict__ bias, float* __restrict__ C,
                      int K, int Nc, int Ncp, int lda) {
    const int lane  = threadIdx.x & 31;
    const int wv    = threadIdx.x >> 5;
    const int l16   = lane & 15;
    const int hlf   = lane >> 4;          // 0 or 1
    const int r0    = blockIdx.y * 16;
    const int col0  = blockIdx.x * 512 + wv * 64;
    const int mbase = hlf * 8;

    f32x8 acc[4];
    #pragma unroll
    for (int t = 0; t < 4; ++t)
        #pragma unroll
        for (int j = 0; j < 8; ++j) acc[t][j] = 0.0f;

    if (ACCUM) {
        #pragma unroll
        for (int t = 0; t < 4; ++t) {
            int n = col0 + 16 * t + l16;
            if (n < Nc) {
                #pragma unroll
                for (int j = 0; j < 8; ++j)
                    acc[t][j] = C[(size_t)(r0 + mbase + j) * Nc + n];
            }
        }
    }

    const float* arow = A + (size_t)(r0 + l16) * lda + hlf * 16;
    const unsigned short* bbase = Bt + hlf * 16;

    for (int k = 0; k < K; k += 32) {
        bf16x16 av = pack_a_frag(arow + k);
        #pragma unroll
        for (int t = 0; t < 4; ++t) {
            int c0 = col0 + 16 * t;            // uniform per wave -> EXEC stays full
            if (c0 < Ncp) {
                int n = c0 + l16;
                u16x16 braw = *(const u16x16*)(bbase + (size_t)n * K + k);
                acc[t] = wmma_bf16(av, __builtin_bit_cast(bf16x16, braw), acc[t]);
            }
        }
    }

    #pragma unroll
    for (int t = 0; t < 4; ++t) {
        int n = col0 + 16 * t + l16;
        if (n < Nc) {
            float bv = bias[n];
            #pragma unroll
            for (int j = 0; j < 8; ++j) {
                float v = acc[t][j] + bv;
                if (ACT == 1) v = fmaxf(v, 0.0f);
                C[(size_t)(r0 + mbase + j) * Nc + n] = v;
            }
        }
    }
}

// ---------------------------------------------------------------------------
// Fused attention score:
//   score[r] = tanh(enc[r,:] @ U_w + Wh[r%N,:] + U_b) . v_w + v_b
// for r in [0, S*N). Block handles 16 rows x full 1024 cols: 8 waves x
// 8 col-tiles kept in accumulators; tanh + v_w dot fused in epilogue so the
// (S,N,H) intermediate never hits memory.  grid = S*N/16, block = 256.
// ---------------------------------------------------------------------------
__global__ __launch_bounds__(256)
void attn_score_kernel(const float* __restrict__ enc, const unsigned short* __restrict__ Ut,
                       const float* __restrict__ Wh, const float* __restrict__ U_b,
                       const float* __restrict__ v_w, const float* __restrict__ v_b,
                       float* __restrict__ score) {
    const int lane  = threadIdx.x & 31;
    const int wv    = threadIdx.x >> 5;
    const int l16   = lane & 15;
    const int hlf   = lane >> 4;
    const int r0    = blockIdx.x * 16;
    const int col0  = wv * 128;
    const int mbase = hlf * 8;

    f32x8 acc[8];
    #pragma unroll
    for (int t = 0; t < 8; ++t)
        #pragma unroll
        for (int j = 0; j < 8; ++j) acc[t][j] = 0.0f;

    const float* arow = enc + (size_t)(r0 + l16) * H_DIM + hlf * 16;
    const unsigned short* bbase = Ut + hlf * 16;

    for (int k = 0; k < H_DIM; k += 32) {
        bf16x16 av = pack_a_frag(arow + k);
        #pragma unroll
        for (int t = 0; t < 8; ++t) {
            int n = col0 + 16 * t + l16;
            u16x16 braw = *(const u16x16*)(bbase + (size_t)n * H_DIM + k);
            acc[t] = wmma_bf16(av, __builtin_bit_cast(bf16x16, braw), acc[t]);
        }
    }

    float partial[8];
    #pragma unroll
    for (int j = 0; j < 8; ++j) partial[j] = 0.0f;

    #pragma unroll
    for (int t = 0; t < 8; ++t) {
        int n = col0 + 16 * t + l16;
        float vw = v_w[n];
        float ub = U_b[n];
        #pragma unroll
        for (int j = 0; j < 8; ++j) {
            int r  = r0 + mbase + j;
            int nb = r & (N_BATCH - 1);
            float val = tanhf(acc[t][j] + Wh[(size_t)nb * H_DIM + n] + ub);
            partial[j] += val * vw;
        }
    }
    // reduce across the 16 lanes of each half-wave (columns of one row)
    #pragma unroll
    for (int off = 1; off < 16; off <<= 1)
        #pragma unroll
        for (int j = 0; j < 8; ++j)
            partial[j] += __shfl_xor(partial[j], off, 32);

    __shared__ float wsum[8][16];
    if (l16 == 0) {                // lanes 0 (rows 0..7) and 16 (rows 8..15)
        #pragma unroll
        for (int j = 0; j < 8; ++j) wsum[wv][mbase + j] = partial[j];
    }
    __syncthreads();
    if (threadIdx.x < 16) {
        float s = v_b[0];
        #pragma unroll
        for (int w = 0; w < 8; ++w) s += wsum[w][threadIdx.x];
        score[r0 + threadIdx.x] = s;
    }
}

// softmax over S per batch column: attn[s,n] = softmax_s(score[s,n])
__global__ __launch_bounds__(128)
void softmax_s_kernel(const float* __restrict__ score, float* __restrict__ attn) {
    int n = blockIdx.x, s = threadIdx.x;
    __shared__ float red[128];
    float v = score[s * N_BATCH + n];
    red[s] = v; __syncthreads();
    for (int off = 64; off > 0; off >>= 1) {
        if (s < off) red[s] = fmaxf(red[s], red[s + off]);
        __syncthreads();
    }
    float mx = red[0]; __syncthreads();
    float e = expf(v - mx);
    red[s] = e; __syncthreads();
    for (int off = 64; off > 0; off >>= 1) {
        if (s < off) red[s] += red[s + off];
        __syncthreads();
    }
    attn[s * N_BATCH + n] = e / red[0];
}

// context[n,:] = sum_s attn[s,n] * enc[s,n,:], written into xc[:, E:E+H]
__global__ __launch_bounds__(256)
void context_kernel(const float* __restrict__ attn, const float* __restrict__ enc,
                    float* __restrict__ xc) {
    int n = blockIdx.x, tid = threadIdx.x;
    float a0 = 0.f, a1 = 0.f, a2 = 0.f, a3 = 0.f;
    for (int s = 0; s < S_ENC; ++s) {
        float a = attn[s * N_BATCH + n];
        const float* er = enc + ((size_t)s * N_BATCH + n) * H_DIM;
        a0 += a * er[tid];
        a1 += a * er[tid + 256];
        a2 += a * er[tid + 512];
        a3 += a * er[tid + 768];
    }
    float* dst = xc + (size_t)n * (E_DIM + H_DIM) + E_DIM;
    dst[tid] = a0; dst[tid + 256] = a1; dst[tid + 512] = a2; dst[tid + 768] = a3;
}

// x = emb[input_ids] -> xc[:, 0:E]
__global__ __launch_bounds__(256)
void gather_x_kernel(const int* __restrict__ ids, const float* __restrict__ emb,
                     float* __restrict__ xc) {
    int n = blockIdx.x;
    size_t row = (size_t)ids[n] * E_DIM;
    for (int j = threadIdx.x; j < E_DIM; j += 256)
        xc[(size_t)n * (E_DIM + H_DIM) + j] = emb[row + j];
}

// GRU gate math + LayerNorm, writes new hidden state (d_out h slot)
__global__ __launch_bounds__(256)
void gru_ln_kernel(const float* __restrict__ gp, const float* __restrict__ cp,
                   const float* __restrict__ hp, const float* __restrict__ hidden,
                   const float* __restrict__ ln_g, const float* __restrict__ ln_b,
                   float* __restrict__ hout) {
    int n = blockIdx.x, tid = threadIdx.x;
    float hv[4];
    float s = 0.f, s2 = 0.f;
    #pragma unroll
    for (int i = 0; i < 4; ++i) {
        int c = tid + 256 * i;
        float z = 1.f / (1.f + expf(-gp[(size_t)n * (2 * H_DIM) + c]));
        float r = 1.f / (1.f + expf(-gp[(size_t)n * (2 * H_DIM) + H_DIM + c]));
        float cand = tanhf(cp[(size_t)n * H_DIM + c] + r * hp[(size_t)n * H_DIM + c]);
        float hprev = hidden[(size_t)n * H_DIM + c];
        float h = (1.f - z) * hprev + z * cand;
        hv[i] = h; s += h; s2 += h * h;
    }
    __shared__ float rs[256], rq[256];
    rs[tid] = s; rq[tid] = s2; __syncthreads();
    for (int off = 128; off > 0; off >>= 1) {
        if (tid < off) { rs[tid] += rs[tid + off]; rq[tid] += rq[tid + off]; }
        __syncthreads();
    }
    float mean = rs[0] * (1.f / H_DIM);
    float var  = rq[0] * (1.f / H_DIM) - mean * mean;
    float inv  = rsqrtf(var + 1e-5f);
    #pragma unroll
    for (int i = 0; i < 4; ++i) {
        int c = tid + 256 * i;
        hout[(size_t)n * H_DIM + c] = (hv[i] - mean) * inv * ln_g[c] + ln_b[c];
    }
}

// in-place log_softmax over V per row
__global__ __launch_bounds__(256)
void logsoftmax_kernel(float* __restrict__ logits) {
    int n = blockIdx.x, tid = threadIdx.x;
    float* row = logits + (size_t)n * V_DIM;
    __shared__ float red[256];
    float mx = -3.4e38f;
    for (int j = tid; j < V_DIM; j += 256) mx = fmaxf(mx, row[j]);
    red[tid] = mx; __syncthreads();
    for (int off = 128; off > 0; off >>= 1) {
        if (tid < off) red[tid] = fmaxf(red[tid], red[tid + off]);
        __syncthreads();
    }
    mx = red[0]; __syncthreads();
    float s = 0.f;
    for (int j = tid; j < V_DIM; j += 256) s += expf(row[j] - mx);
    red[tid] = s; __syncthreads();
    for (int off = 128; off > 0; off >>= 1) {
        if (tid < off) red[tid] += red[tid + off];
        __syncthreads();
    }
    float lse = mx + logf(red[0]);
    for (int j = tid; j < V_DIM; j += 256) row[j] -= lse;
}

// ---------------------------------------------------------------------------
extern "C" void kernel_launch(void* const* d_in, const int* in_sizes, int n_in,
                              void* d_out, int out_size, void* d_ws, size_t ws_size,
                              hipStream_t stream) {
    (void)in_sizes; (void)n_in; (void)out_size; (void)ws_size;

    const int*   ids    = (const int*)  d_in[0];
    const float* hidden = (const float*)d_in[1];
    const float* enc    = (const float*)d_in[2];
    const float* emb    = (const float*)d_in[3];
    const float* W_w    = (const float*)d_in[4];
    const float* W_b    = (const float*)d_in[5];
    const float* U_w    = (const float*)d_in[6];
    const float* U_b    = (const float*)d_in[7];
    const float* v_w    = (const float*)d_in[8];
    const float* v_b    = (const float*)d_in[9];
    const float* ih_w   = (const float*)d_in[10];
    const float* ih_b   = (const float*)d_in[11];
    const float* hh_w   = (const float*)d_in[12];
    const float* hh_b   = (const float*)d_in[13];
    const float* cand_w = (const float*)d_in[14];
    const float* cand_b = (const float*)d_in[15];
    const float* hhc_w  = (const float*)d_in[16];
    const float* hhc_b  = (const float*)d_in[17];
    const float* comb_w = (const float*)d_in[18];
    const float* comb_b = (const float*)d_in[19];
    const float* ln_g   = (const float*)d_in[20];
    const float* ln_b   = (const float*)d_in[21];
    const float* out1_w = (const float*)d_in[22];
    const float* out1_b = (const float*)d_in[23];
    const float* out2_w = (const float*)d_in[24];
    const float* out2_b = (const float*)d_in[25];

    // d_out layout: logits (N,V) | h (N,H) | attn (S,N,1)
    float* logits = (float*)d_out;
    float* hout   = logits + (size_t)N_BATCH * V_DIM;
    float* attn   = hout   + (size_t)N_BATCH * H_DIM;

    // workspace carve-out (256B aligned)
    char* base = (char*)d_ws;
    size_t off = 0;
    auto alloc = [&](size_t bytes) -> void* {
        void* r = base + off;
        off = (off + bytes + 255) & ~(size_t)255;
        return r;
    };
    unsigned short* WtW   = (unsigned short*)alloc((size_t)H_DIM * H_DIM * 2);
    unsigned short* Ut    = (unsigned short*)alloc((size_t)H_DIM * H_DIM * 2);
    unsigned short* combT = (unsigned short*)alloc((size_t)E_DIM * (E_DIM + H_DIM) * 2);
    unsigned short* ihT   = (unsigned short*)alloc((size_t)(2 * H_DIM) * E_DIM * 2);
    unsigned short* hhT   = (unsigned short*)alloc((size_t)(2 * H_DIM) * H_DIM * 2);
    unsigned short* candT = (unsigned short*)alloc((size_t)H_DIM * E_DIM * 2);
    unsigned short* hhcT  = (unsigned short*)alloc((size_t)H_DIM * H_DIM * 2);
    unsigned short* out1T = (unsigned short*)alloc((size_t)H_DIM * H_DIM * 2);
    unsigned short* out2T = (unsigned short*)alloc((size_t)V_PAD * H_DIM * 2);
    float* Wh    = (float*)alloc((size_t)N_BATCH * H_DIM * 4);
    float* score = (float*)alloc((size_t)S_ENC * N_BATCH * 4);
    float* xc    = (float*)alloc((size_t)N_BATCH * (E_DIM + H_DIM) * 4);
    float* g     = (float*)alloc((size_t)N_BATCH * E_DIM * 4);
    float* gp    = (float*)alloc((size_t)N_BATCH * 2 * H_DIM * 4);
    float* cp    = (float*)alloc((size_t)N_BATCH * H_DIM * 4);
    float* hp    = (float*)alloc((size_t)N_BATCH * H_DIM * 4);
    float* r1    = (float*)alloc((size_t)N_BATCH * H_DIM * 4);

    dim3 tb(32, 8);
    // weight transpose + bf16 convert (every call; no cross-call state allowed)
    convT_kernel<<<dim3(H_DIM/32,        H_DIM/32),          tb, 0, stream>>>(W_w,    WtW,   H_DIM,         H_DIM,       H_DIM);
    convT_kernel<<<dim3(H_DIM/32,        H_DIM/32),          tb, 0, stream>>>(U_w,    Ut,    H_DIM,         H_DIM,       H_DIM);
    convT_kernel<<<dim3(E_DIM/32,        (E_DIM+H_DIM)/32),  tb, 0, stream>>>(comb_w, combT, E_DIM + H_DIM, E_DIM,       E_DIM);
    convT_kernel<<<dim3((2*H_DIM)/32,    E_DIM/32),          tb, 0, stream>>>(ih_w,   ihT,   E_DIM,         2 * H_DIM,   2 * H_DIM);
    convT_kernel<<<dim3((2*H_DIM)/32,    H_DIM/32),          tb, 0, stream>>>(hh_w,   hhT,   H_DIM,         2 * H_DIM,   2 * H_DIM);
    convT_kernel<<<dim3(H_DIM/32,        E_DIM/32),          tb, 0, stream>>>(cand_w, candT, E_DIM,         H_DIM,       H_DIM);
    convT_kernel<<<dim3(H_DIM/32,        H_DIM/32),          tb, 0, stream>>>(hhc_w,  hhcT,  H_DIM,         H_DIM,       H_DIM);
    convT_kernel<<<dim3(H_DIM/32,        H_DIM/32),          tb, 0, stream>>>(out1_w, out1T, H_DIM,         H_DIM,       H_DIM);
    convT_kernel<<<dim3(V_PAD/32,        H_DIM/32),          tb, 0, stream>>>(out2_w, out2T, H_DIM,         V_DIM,       V_PAD);

    // x = emb[ids] -> xc[:, :E]
    gather_x_kernel<<<N_BATCH, 256, 0, stream>>>(ids, emb, xc);

    // Wh = hidden @ W_w + W_b
    gemm_bf16_kernel<0, false><<<dim3(2, N_BATCH/16), 256, 0, stream>>>(
        hidden, WtW, W_b, Wh, H_DIM, H_DIM, H_DIM, H_DIM);

    // score[s,n] = tanh(enc @ U_w + Wh + U_b) . v_w + v_b   (fused, 68.7 GFLOP)
    attn_score_kernel<<<(S_ENC * N_BATCH) / 16, 256, 0, stream>>>(
        enc, Ut, Wh, U_b, v_w, v_b, score);

    // attn = softmax_s(score)  -> written straight into d_out attn slot
    softmax_s_kernel<<<N_BATCH, S_ENC, 0, stream>>>(score, attn);

    // context -> xc[:, E:]
    context_kernel<<<N_BATCH, 256, 0, stream>>>(attn, enc, xc);

    // g = relu(xc @ comb_w + comb_b)
    gemm_bf16_kernel<1, false><<<dim3(1, N_BATCH/16), 256, 0, stream>>>(
        xc, combT, comb_b, g, E_DIM + H_DIM, E_DIM, E_DIM, E_DIM + H_DIM);

    // gp = g @ ih_w + ih_b ; gp += hidden @ hh_w + hh_b
    gemm_bf16_kernel<0, false><<<dim3(4, N_BATCH/16), 256, 0, stream>>>(
        g, ihT, ih_b, gp, E_DIM, 2 * H_DIM, 2 * H_DIM, E_DIM);
    gemm_bf16_kernel<0, true><<<dim3(4, N_BATCH/16), 256, 0, stream>>>(
        hidden, hhT, hh_b, gp, H_DIM, 2 * H_DIM, 2 * H_DIM, H_DIM);

    // cp = g @ cand_w + cand_b ; hp = hidden @ hhc_w + hhc_b
    gemm_bf16_kernel<0, false><<<dim3(2, N_BATCH/16), 256, 0, stream>>>(
        g, candT, cand_b, cp, E_DIM, H_DIM, H_DIM, E_DIM);
    gemm_bf16_kernel<0, false><<<dim3(2, N_BATCH/16), 256, 0, stream>>>(
        hidden, hhcT, hhc_b, hp, H_DIM, H_DIM, H_DIM, H_DIM);

    // GRU update + LayerNorm -> h (d_out)
    gru_ln_kernel<<<N_BATCH, 256, 0, stream>>>(gp, cp, hp, hidden, ln_g, ln_b, hout);

    // r1 = relu(h @ out1_w + out1_b)
    gemm_bf16_kernel<1, false><<<dim3(2, N_BATCH/16), 256, 0, stream>>>(
        hout, out1T, out1_b, r1, H_DIM, H_DIM, H_DIM, H_DIM);

    // logits = r1 @ out2_w + out2_b   (26.3 GFLOP, V padded to 50272)
    gemm_bf16_kernel<0, false><<<dim3((V_PAD + 511) / 512, N_BATCH/16), 256, 0, stream>>>(
        r1, out2T, out2_b, logits, H_DIM, V_DIM, V_PAD, H_DIM);

    // log_softmax in place over V
    logsoftmax_kernel<<<N_BATCH, 256, 0, stream>>>(logits);
}
